// minus_attention_23854248362445
// MI455X (gfx1250) — matrix operand order repaired and checked by reading, not compiled, not run on hardware
//
#include <hip/hip_runtime.h>
#include <math.h>

// Problem constants (match reference)
#define BATCH 4
#define LXD   384
#define LYD   384
#define HD    256   // hidden
#define FD    512   // feature = 2H

typedef __attribute__((ext_vector_type(16))) __bf16 v16bf;
typedef __attribute__((ext_vector_type(8)))  float  v8f;

// Native CDNA5 V_TANH_F32 when available (trans op, co-executes with VALU).
__device__ __forceinline__ float fast_tanh(float a) {
#if __has_builtin(__builtin_amdgcn_tanhf)
  return __builtin_amdgcn_tanhf(a);
#else
  return tanhf(a);
#endif
}

// Convert 8 consecutive fp32 into bf16 hi/lo split, placed at [base, base+8) of v16bf pair.
__device__ __forceinline__ void cvt8(const float* __restrict__ p, v16bf& hi, v16bf& lo, int base) {
  float4 q0 = *reinterpret_cast<const float4*>(p);
  float4 q1 = *reinterpret_cast<const float4*>(p + 4);
  float f[8] = {q0.x, q0.y, q0.z, q0.w, q1.x, q1.y, q1.z, q1.w};
#pragma unroll
  for (int i = 0; i < 8; ++i) {
    __bf16 h = (__bf16)f[i];
    hi[base + i] = h;
    lo[base + i] = (__bf16)(f[i] - (float)h);
  }
}

// bf16x3 compensated accumulate: acc += (Ah+Al)*(Bh+Bl) dropping Al*Bl
__device__ __forceinline__ v8f wmma_bf16x3(v16bf ah, v16bf al, v16bf bh, v16bf bl, v8f acc) {
  acc = __builtin_amdgcn_wmma_f32_16x16x32_bf16(false, ah, false, bh, (short)0, acc, false, false);
  acc = __builtin_amdgcn_wmma_f32_16x16x32_bf16(false, ah, false, bl, (short)0, acc, false, false);
  acc = __builtin_amdgcn_wmma_f32_16x16x32_bf16(false, al, false, bh, (short)0, acc, false, false);
  return acc;
}

// ---------------------------------------------------------------------------
// Kernel 1: Xp = x @ Wm^T, Yp = y @ Wm^T.
// x,y: (1536, 512) row-major; Wm: (256, 512) row-major (N x K, i.e. B^T).
// One wave per 16x16 output tile. Waves 0..1535 -> x, 1536..3071 -> y.
// ---------------------------------------------------------------------------
__global__ void __launch_bounds__(256) proj_kernel(
    const float* __restrict__ x, const float* __restrict__ y,
    const float* __restrict__ Wm,
    float* __restrict__ Xp, float* __restrict__ Yp) {
  const int wave = (blockIdx.x * blockDim.x + threadIdx.x) >> 5;   // 0..3071
  const int lane = threadIdx.x & 31;
  const int src   = wave >= 1536;           // 0: x, 1: y
  const int local = wave - src * 1536;
  const int mtile = local >> 4;             // 0..95  (M = 1536 rows)
  const int ntile = local & 15;             // 0..15  (N = 256 cols)
  const int m0 = mtile << 4, n0 = ntile << 4;

  const float* __restrict__ in  = src ? y  : x;
  float* __restrict__       out = src ? Yp : Xp;

  const int half = lane >> 4;               // which half-wave
  const int l15  = lane & 15;
  const float* arow = in + (size_t)(m0 + l15) * FD;   // A row (M x K layout)
  const float* brow = Wm + (size_t)(n0 + l15) * FD;   // B column = Wm row (contiguous)

  v8f acc = {};
  for (int k0 = 0; k0 < FD; k0 += 32) {
    v16bf ah, al, bh, bl;
    // A 16x32 bf16, contiguous k order chosen here (B matches): lanes<16 hold K k0..k0+15 of row m; lanes>=16 hold k0+16..k0+31.
    const int kbA = k0 + half * 16;
    cvt8(arow + kbA,      ah, al, 0);
    cvt8(arow + kbA + 8,  ah, al, 8);
    // B 32x16 bf16, same contiguous k order as A: lanes<16 hold K k0..k0+15 of column n; lanes>=16 hold k0+16..k0+31.
    const int kbB = k0 + half * 16;
    cvt8(brow + kbB,     bh, bl, 0);
    cvt8(brow + kbB + 8, bh, bl, 8);
    acc = wmma_bf16x3(ah, al, bh, bl, acc);
  }
  // C/D layout: VGPR r -> (M = m0 + r + 8*half, N = n0 + l15)
#pragma unroll
  for (int r = 0; r < 8; ++r) {
    out[(size_t)(m0 + r + half * 8) * HD + (n0 + l15)] = acc[r];
  }
}

// ---------------------------------------------------------------------------
// Kernel 2: fused score + softmax.
// One block per (b, j): s[l] = sum_h vm[h]*tanh(Xp[b,l,h]-Yp[b,j,h]); softmax over l.
// 8 waves; each wave owns one l at a time. Per-wave Yp/vm fragments are held in
// registers (l-invariant); Xp row read as two b128 loads per lane; wave32
// shuffle reduce; native v_tanh_f32 where available.
// ---------------------------------------------------------------------------
__global__ void __launch_bounds__(256) score_softmax_kernel(
    const float* __restrict__ Xp, const float* __restrict__ Yp,
    const float* __restrict__ vm, float* __restrict__ ait) {
  __shared__ __align__(16) float sY[HD];
  __shared__ __align__(16) float sV[HD];
  __shared__ float sS[LXD];
  __shared__ float red[256];

  const int t = threadIdx.x;
  const int b = blockIdx.x / LYD;
  const int j = blockIdx.x - b * LYD;

  sY[t] = Yp[((size_t)b * LYD + j) * HD + t];
  sV[t] = vm[t];
  __syncthreads();

  const int wave = t >> 5, lane = t & 31;

  // Hoist the per-lane Yp/vm fragments out of the l loop (h = 4*lane + {0..3}
  // and 128 + 4*lane + {0..3}).
  const float4* sY4 = reinterpret_cast<const float4*>(sY);
  const float4* sV4 = reinterpret_cast<const float4*>(sV);
  const float4 ya = sY4[lane], yb = sY4[lane + 32];
  const float4 va = sV4[lane], vb = sV4[lane + 32];

  const float* xbase = Xp + (size_t)b * LXD * HD;
  for (int l = wave; l < LXD; l += 8) {
    const float4* xr4 = reinterpret_cast<const float4*>(xbase + (size_t)l * HD);
    if (l + 8 < LXD)  // prefetch this wave's next row (global_prefetch_b8)
      __builtin_prefetch(xbase + (size_t)(l + 8) * HD + 4 * lane, 0, 0);
    const float4 xa = xr4[lane];
    const float4 xb = xr4[lane + 32];
    float s;
    s  = va.x * fast_tanh(xa.x - ya.x);
    s += va.y * fast_tanh(xa.y - ya.y);
    s += va.z * fast_tanh(xa.z - ya.z);
    s += va.w * fast_tanh(xa.w - ya.w);
    s += vb.x * fast_tanh(xb.x - yb.x);
    s += vb.y * fast_tanh(xb.y - yb.y);
    s += vb.z * fast_tanh(xb.z - yb.z);
    s += vb.w * fast_tanh(xb.w - yb.w);
#pragma unroll
    for (int off = 16; off > 0; off >>= 1) s += __shfl_xor(s, off, 32);
    if (lane == 0) sS[l] = s;
  }
  __syncthreads();

  // softmax over sS[0..383]
  float m = -INFINITY;
  for (int l = t; l < LXD; l += 256) m = fmaxf(m, sS[l]);
  red[t] = m;
  __syncthreads();
  for (int st = 128; st > 0; st >>= 1) {
    if (t < st) red[t] = fmaxf(red[t], red[t + st]);
    __syncthreads();
  }
  m = red[0];
  __syncthreads();

  float sum = 0.f;
  for (int l = t; l < LXD; l += 256) {
    const float e = __expf(sS[l] - m);
    sS[l] = e;
    sum += e;
  }
  red[t] = sum;
  __syncthreads();
  for (int st = 128; st > 0; st >>= 1) {
    if (t < st) red[t] += red[t + st];
    __syncthreads();
  }
  const float inv = 1.0f / red[0];

  float* arow = ait + ((size_t)b * LYD + j) * LXD;
  for (int l = t; l < LXD; l += 256) arow[l] = sS[l] * inv;
}

// ---------------------------------------------------------------------------
// Kernel 3: qtm[b] = ait[b] @ x[b].  A: (384 x 384) row-major, B: (384 x 512)
// row-major (K x N). One wave per 16x16 tile; 4*24*32 = 3072 waves.
// ---------------------------------------------------------------------------
__global__ void __launch_bounds__(256) out_gemm_kernel(
    const float* __restrict__ ait, const float* __restrict__ x,
    float* __restrict__ qtm) {
  const int wave = (blockIdx.x * blockDim.x + threadIdx.x) >> 5;   // 0..3071
  const int lane = threadIdx.x & 31;
  const int b = wave / 768;
  const int local = wave - b * 768;
  const int mtile = local >> 5;     // 0..23  (M = 384)
  const int ntile = local & 31;     // 0..31  (N = 512)
  const int m0 = mtile << 4, n0 = ntile << 4;

  const float* A  = ait + (size_t)b * LYD * LXD;
  const float* Bx = x   + (size_t)b * LXD * FD;
  float* C        = qtm + (size_t)b * LYD * FD;

  const int half = lane >> 4;
  const int l15  = lane & 15;
  const float* arow = A + (size_t)(m0 + l15) * LXD;
  const int col = n0 + l15;

  v8f acc = {};
  for (int k0 = 0; k0 < LXD; k0 += 32) {
    v16bf ah, al, bh, bl;
    const int kbA = k0 + half * 16;
    cvt8(arow + kbA,      ah, al, 0);
    cvt8(arow + kbA + 8,  ah, al, 8);
    const int kbB = k0 + half * 16;
#pragma unroll
    for (int e = 0; e < 16; ++e) {           // B stored K x N: per-lane strided scalar loads
      const float v = Bx[(size_t)(kbB + e) * FD + col];
      const __bf16 h = (__bf16)v;
      bh[e] = h;
      bl[e] = (__bf16)(v - (float)h);
    }
    acc = wmma_bf16x3(ah, al, bh, bl, acc);
  }
#pragma unroll
  for (int r = 0; r < 8; ++r) {
    C[(size_t)(m0 + r + half * 8) * FD + col] = acc[r];
  }
}

// ---------------------------------------------------------------------------
extern "C" void kernel_launch(void* const* d_in, const int* in_sizes, int n_in,
                              void* d_out, int out_size, void* d_ws, size_t ws_size,
                              hipStream_t stream) {
  const float* x  = (const float*)d_in[0];   // (4, 384, 512)
  const float* y  = (const float*)d_in[1];   // (4, 384, 512)
  const float* Wm = (const float*)d_in[2];   // (256, 512)
  const float* vm = (const float*)d_in[3];   // (256,)
  float* out = (float*)d_out;                // (4, 384, 512)

  float* Xp  = (float*)d_ws;                       // 4*384*256 f32 = 1.5 MB
  float* Yp  = Xp + (size_t)BATCH * LXD * HD;      // 1.5 MB
  float* ait = Yp + (size_t)BATCH * LYD * HD;      // 4*384*384 f32 = 2.25 MB

  // 3072 tile-waves / 8 waves per block = 384 blocks
  proj_kernel<<<384, 256, 0, stream>>>(x, y, Wm, Xp, Yp);
  // one block per (b, j) row
  score_softmax_kernel<<<BATCH * LYD, 256, 0, stream>>>(Xp, Yp, vm, ait);
  // 3072 tile-waves / 8 = 384 blocks
  out_gemm_kernel<<<384, 256, 0, stream>>>(ait, x, out);
}
